// GeometricAttentionGate_46523085750369
// MI455X (gfx1250) — matrix operand order repaired and checked by reading, not compiled, not run on hardware
//
#include <hip/hip_runtime.h>
#include <math.h>

typedef __attribute__((ext_vector_type(16))) _Float16 v16h;
typedef __attribute__((ext_vector_type(2)))  _Float16 h2v;
typedef __attribute__((ext_vector_type(8)))  float    v8f;
typedef __attribute__((ext_vector_type(4)))  float    f4;
typedef __attribute__((ext_vector_type(2)))  unsigned int u32x2;

#define B_   8192
#define D_   512
#define S_   5
#define C_   2048
#define H_   4
#define HD_  128

#define BM 64
#define BN 128
#define BK 32
#define LSA_STRIDE 36    // halves; 72B rows: 8B-aligned, few bank collisions
#define LSB_STRIDE 132   // halves; 264B rows: 8B-aligned, conflict-free frag reads
#define LSK_STRIDE 132   // f32

union F16x16 { v16h h; unsigned int u[8]; };

// pack 4 f32 -> 2 dwords of f16 (RNE, lowers to v_cvt + v_pack_b32_f16)
__device__ inline u32x2 pack4(float a, float b, float c, float d) {
  h2v lo = { (_Float16)a, (_Float16)b };
  h2v hi = { (_Float16)c, (_Float16)d };
  u32x2 r;
  r.x = __builtin_bit_cast(unsigned int, lo);
  r.y = __builtin_bit_cast(unsigned int, hi);
  return r;
}

// ---- A fragment: 16x32 f16, M = lane&15, K pairs per ISA layout ----
__device__ inline void load_afrag(const _Float16* lsA, int mrow, int lane, F16x16& af) {
  const int M = mrow + (lane & 15);
  const int kbase = (lane >> 4) << 3;
  const _Float16* rp = lsA + M * LSA_STRIDE;
#pragma unroll
  for (int e2 = 0; e2 < 8; ++e2) {
    const int K = kbase + 2 * e2 + ((e2 >= 4) ? 8 : 0);
    af.u[e2] = *(const unsigned int*)&rp[K];
  }
}

// ---- B fragment: 32x16 f16, N = lane&15, K pairs per ISA layout ----
__device__ inline void load_bfrag(const _Float16* lsB, int ncol, int lane, F16x16& bf) {
  const _Float16* rp = lsB + ncol + (lane & 15);
  const int kbase = (lane >> 4) << 3;
#pragma unroll
  for (int e2 = 0; e2 < 8; ++e2) {
    const int K = kbase + 2 * e2 + ((e2 >= 4) ? 8 : 0);
    bf.h[2 * e2] = rp[K * LSB_STRIDE];
    bf.h[2 * e2 + 1] = rp[(K + 1) * LSB_STRIDE];
  }
}

// =====================================================================
// Kernel 1: Q = features @ Wq + bq    (f16 WMMA, f32 accumulate)
// =====================================================================
__global__ __launch_bounds__(256) void qgemm_kernel(
    const float* __restrict__ feat, const float* __restrict__ Wq,
    const float* __restrict__ bq, float* __restrict__ Q) {
  __shared__ __align__(16) _Float16 lsA[BM * LSA_STRIDE];
  __shared__ __align__(16) _Float16 lsB[BK * LSB_STRIDE];

  const int bM = blockIdx.x * BM;
  const int bN = blockIdx.y * BN;
  const int tid = threadIdx.x;
  const int wave = tid >> 5, lane = tid & 31;
  const int rg = wave >> 1, cg = wave & 1;
  const int mrow = rg * 16;

  v8f zero = {};
  v8f acc[4] = {zero, zero, zero, zero};

  for (int k0 = 0; k0 < D_; k0 += BK) {
#pragma unroll
    for (int i = 0; i < (BM * BK) / (4 * 256); ++i) {     // stage A: 64x32, f4 wide
      const int lin = tid + 256 * i;
      const int r = lin >> 3, c4 = (lin & 7) * 4;         // 8 quads per row
      const f4 v = *(const f4*)&feat[(size_t)(bM + r) * D_ + k0 + c4];
      *(u32x2*)&lsA[r * LSA_STRIDE + c4] = pack4(v.x, v.y, v.z, v.w);
    }
#pragma unroll
    for (int i = 0; i < (BK * BN) / (4 * 256); ++i) {     // stage B: 32x128, f4 wide
      const int lin = tid + 256 * i;
      const int r = lin >> 5, c4 = (lin & 31) * 4;        // 32 quads per row
      const f4 v = *(const f4*)&Wq[(size_t)(k0 + r) * D_ + bN + c4];
      *(u32x2*)&lsB[r * LSB_STRIDE + c4] = pack4(v.x, v.y, v.z, v.w);
    }
    __syncthreads();

    F16x16 af;
    load_afrag(lsA, mrow, lane, af);
#pragma unroll
    for (int tt = 0; tt < 4; ++tt) {
      F16x16 bf;
      load_bfrag(lsB, cg * 64 + tt * 16, lane, bf);
      acc[tt] = __builtin_amdgcn_wmma_f32_16x16x32_f16(
          false, af.h, false, bf.h, (short)0, acc[tt], false, false);
    }
    __syncthreads();
  }

#pragma unroll
  for (int tt = 0; tt < 4; ++tt)
#pragma unroll
    for (int r = 0; r < 8; ++r) {
      const int lrow = mrow + r + 8 * (lane >> 4);
      const int lcol = cg * 64 + tt * 16 + (lane & 15);
      Q[(size_t)(bM + lrow) * D_ + bN + lcol] = acc[tt][r] + bq[bN + lcol];
    }
}

// =====================================================================
// Kernel 2: K-tile GEMM fused with scores[b,h,s] = Q[b,h]·K[s,b,h]/sqrt(hd)
// (BN=128 == head width; K never written to HBM)
// =====================================================================
__global__ __launch_bounds__(256) void kscore_kernel(
    const float* __restrict__ sfeat, const float* __restrict__ semb,
    const float* __restrict__ Wk, const float* __restrict__ bk,
    const float* __restrict__ Q, float* __restrict__ scores) {
  __shared__ __align__(16) _Float16 lsA[BM * LSA_STRIDE];
  __shared__ __align__(16) _Float16 lsB[BK * LSB_STRIDE];
  __shared__ float lsK[BM * LSK_STRIDE];

  const int s  = blockIdx.z;
  const int bM = blockIdx.x * BM;
  const int bN = blockIdx.y * BN;
  const int tid = threadIdx.x;
  const int wave = tid >> 5, lane = tid & 31;
  const int rg = wave >> 1, cg = wave & 1;
  const int mrow = rg * 16;

  const float* A  = sfeat + (size_t)s * B_ * D_;
  const float* se = semb + s * D_;

  v8f zero = {};
  v8f acc[4] = {zero, zero, zero, zero};

  for (int k0 = 0; k0 < D_; k0 += BK) {
#pragma unroll
    for (int i = 0; i < (BM * BK) / (4 * 256); ++i) {
      const int lin = tid + 256 * i;
      const int r = lin >> 3, c4 = (lin & 7) * 4;
      const f4 v = *(const f4*)&A[(size_t)(bM + r) * D_ + k0 + c4];
      const f4 e = *(const f4*)&se[k0 + c4];
      *(u32x2*)&lsA[r * LSA_STRIDE + c4] =
          pack4(v.x + e.x, v.y + e.y, v.z + e.z, v.w + e.w);
    }
#pragma unroll
    for (int i = 0; i < (BK * BN) / (4 * 256); ++i) {
      const int lin = tid + 256 * i;
      const int r = lin >> 5, c4 = (lin & 31) * 4;
      const f4 v = *(const f4*)&Wk[(size_t)(k0 + r) * D_ + bN + c4];
      *(u32x2*)&lsB[r * LSB_STRIDE + c4] = pack4(v.x, v.y, v.z, v.w);
    }
    __syncthreads();

    F16x16 af;
    load_afrag(lsA, mrow, lane, af);
#pragma unroll
    for (int tt = 0; tt < 4; ++tt) {
      F16x16 bf;
      load_bfrag(lsB, cg * 64 + tt * 16, lane, bf);
      acc[tt] = __builtin_amdgcn_wmma_f32_16x16x32_f16(
          false, af.h, false, bf.h, (short)0, acc[tt], false, false);
    }
    __syncthreads();
  }

  // K tile (with bias) -> LDS
#pragma unroll
  for (int tt = 0; tt < 4; ++tt)
#pragma unroll
    for (int r = 0; r < 8; ++r) {
      const int lrow = mrow + r + 8 * (lane >> 4);
      const int lcol = cg * 64 + tt * 16 + (lane & 15);
      lsK[lrow * LSK_STRIDE + lcol] = acc[tt][r] + bk[bN + lcol];
    }
  __syncthreads();

  // scores: 4 threads per row, interleaved columns (conflict-free)
  const int row = tid >> 2, q = tid & 3;
  const float* qrow = Q + (size_t)(bM + row) * D_ + bN;
  float sum = 0.f;
#pragma unroll
  for (int j = 0; j < 32; ++j) {
    const int col = q + 4 * j;
    sum += lsK[row * LSK_STRIDE + col] * qrow[col];
  }
  sum += __shfl_xor(sum, 1);
  sum += __shfl_xor(sum, 2);
  if (q == 0) {
    const int h = bN >> 7;   // head index (head width 128 == BN)
    scores[(size_t)(bM + row) * (H_ * S_) + h * S_ + s] =
        sum * 0.08838834764831845f;  // 1/sqrt(128)
  }
}

// =====================================================================
// Kernel 3: per-row scalars ff=|f|^2, ss=|sf|^2, fs=f·sf (one wave/row)
// =====================================================================
__global__ __launch_bounds__(256) void dots_kernel(
    const float* __restrict__ feat, const float* __restrict__ sfeat,
    float* __restrict__ ff, float* __restrict__ ss, float* __restrict__ fs) {
  const int wave = threadIdx.x >> 5, lane = threadIdx.x & 31;
  const int b = blockIdx.x * 8 + wave;
  const f4* fr = (const f4*)(feat + (size_t)b * D_);

  f4 f[4];
  float accf = 0.f;
#pragma unroll
  for (int i = 0; i < 4; ++i) {
    f[i] = fr[lane + 32 * i];
    accf += f[i].x * f[i].x + f[i].y * f[i].y + f[i].z * f[i].z + f[i].w * f[i].w;
  }
#pragma unroll
  for (int m = 16; m >= 1; m >>= 1) accf += __shfl_xor(accf, m);
  if (lane == 0) ff[b] = accf;

  for (int s = 0; s < S_; ++s) {
    const f4* sr = (const f4*)(sfeat + ((size_t)s * B_ + b) * D_);
    float a_ss = 0.f, a_fs = 0.f;
#pragma unroll
    for (int i = 0; i < 4; ++i) {
      const f4 v = sr[lane + 32 * i];
      a_ss += v.x * v.x + v.y * v.y + v.z * v.z + v.w * v.w;
      a_fs += v.x * f[i].x + v.y * f[i].y + v.z * f[i].z + v.w * f[i].w;
    }
#pragma unroll
    for (int m = 16; m >= 1; m >>= 1) {
      a_ss += __shfl_xor(a_ss, m);
      a_fs += __shfl_xor(a_fs, m);
    }
    if (lane == 0) { ss[s * B_ + b] = a_ss; fs[s * B_ + b] = a_fs; }
  }
}

// =====================================================================
// Kernel 4: fuse the three attentions -> combined [B,S]
// Cayley-Menger collapses to (18*ff + (18-8√2)*ss - (8+4√2)*fs)/25
// =====================================================================
__global__ __launch_bounds__(256) void fuse_kernel(
    const float* __restrict__ scores, const float* __restrict__ ff,
    const float* __restrict__ ss, const float* __restrict__ fs,
    const float* __restrict__ temp, const float* __restrict__ attnw,
    float* __restrict__ comb) {
  const int b = blockIdx.x * 256 + threadIdx.x;
  if (b >= B_) return;
  const float t = fabsf(temp[0]);
  const float inv_t = 1.f / t;

  // --- MHA: softmax over s per head, mean over heads ---
  float mha[S_] = {0.f, 0.f, 0.f, 0.f, 0.f};
  const float* sb = scores + (size_t)b * H_ * S_;
#pragma unroll
  for (int h = 0; h < H_; ++h) {
    float x[S_], xm = -1e30f;
#pragma unroll
    for (int s2 = 0; s2 < S_; ++s2) { x[s2] = sb[h * S_ + s2] * inv_t; xm = fmaxf(xm, x[s2]); }
    float sum = 0.f;
#pragma unroll
    for (int s2 = 0; s2 < S_; ++s2) { x[s2] = expf(x[s2] - xm); sum += x[s2]; }
    const float inv = 0.25f / sum;
#pragma unroll
    for (int s2 = 0; s2 < S_; ++s2) mha[s2] += x[s2] * inv;
  }

  const float ffb = ff[b];
  const float nf = fmaxf(sqrtf(ffb), 1e-12f);

  float g[S_], gm = -1e30f;
  float v[S_], vm = -1e30f;
#pragma unroll
  for (int s2 = 0; s2 < S_; ++s2) {
    const float ssv = ss[s2 * B_ + b];
    const float fsv = fs[s2 * B_ + b];
    float cs = fsv / (nf * fmaxf(sqrtf(ssv), 1e-12f));
    cs = fminf(fmaxf(cs, -1.f + 1e-7f), 1.f - 1e-7f);
    g[s2] = expf(-acosf(cs) * inv_t);          // input to geo softmax
    gm = fmaxf(gm, g[s2]);
    v[s2] = (18.f * ffb + 6.686291501015239f * ssv - 13.65685424949238f * fsv)
            * 0.04f * inv_t;                   // Cayley-Menger closed form / t
    vm = fmaxf(vm, v[s2]);
  }
  float gsum = 0.f, vsum = 0.f;
#pragma unroll
  for (int s2 = 0; s2 < S_; ++s2) {
    g[s2] = expf(g[s2] - gm); gsum += g[s2];
    v[s2] = expf(v[s2] - vm); vsum += v[s2];
  }

  float w0 = attnw[0], w1 = attnw[1], w2 = attnw[2];
  const float wm = fmaxf(w0, fmaxf(w1, w2));
  w0 = expf(w0 - wm); w1 = expf(w1 - wm); w2 = expf(w2 - wm);
  const float wsum = w0 + w1 + w2;
  w0 /= wsum; w1 /= wsum; w2 /= wsum;

  float cmb[S_], csum = 0.f;
#pragma unroll
  for (int s2 = 0; s2 < S_; ++s2) {
    cmb[s2] = w0 * mha[s2] + w1 * (g[s2] / gsum) + w2 * (v[s2] / vsum);
    csum += cmb[s2];
  }
  const float inv = 1.f / csum;
#pragma unroll
  for (int s2 = 0; s2 < S_; ++s2) comb[(size_t)b * S_ + s2] = cmb[s2] * inv;
}

// =====================================================================
// Kernel 5: combined_logits[b,c] = sum_s comb[b,s] * logits[s,b,c]
// b128 nontemporal stream of the 335 MB logits tensor
// =====================================================================
__global__ __launch_bounds__(256) void logits_kernel(
    const float* __restrict__ logits, const float* __restrict__ comb,
    float* __restrict__ out) {
  const size_t idx4 = (size_t)blockIdx.x * 256 + threadIdx.x;
  const int b  = (int)(idx4 >> 9);            // C_/4 = 512 quads per row
  const int c4 = (int)(idx4 & 511) * 4;
  f4 acc = {0.f, 0.f, 0.f, 0.f};
#pragma unroll
  for (int s = 0; s < S_; ++s) {
    const float w = comb[(size_t)b * S_ + s];
    const f4 v = __builtin_nontemporal_load(
        (const f4*)&logits[((size_t)s * B_ + b) * C_ + c4]);
    acc.x += w * v.x; acc.y += w * v.y; acc.z += w * v.z; acc.w += w * v.w;
  }
  __builtin_nontemporal_store(acc, (f4*)&out[(size_t)b * C_ + c4]);
}

extern "C" void kernel_launch(void* const* d_in, const int* in_sizes, int n_in,
                              void* d_out, int out_size, void* d_ws, size_t ws_size,
                              hipStream_t stream) {
  const float* feat   = (const float*)d_in[0];
  const float* logits = (const float*)d_in[1];
  const float* sfeat  = (const float*)d_in[2];
  const float* Wq     = (const float*)d_in[3];
  const float* bq     = (const float*)d_in[4];
  const float* Wk     = (const float*)d_in[5];
  const float* bk     = (const float*)d_in[6];
  // d_in[7] = Wv, d_in[8] = bv : computed-but-unused in the reference
  const float* semb   = (const float*)d_in[9];
  const float* temp   = (const float*)d_in[10];
  const float* attnw  = (const float*)d_in[11];

  float* out = (float*)d_out;                       // [B*C] ++ [B*S]
  float* ws  = (float*)d_ws;
  float* Q      = ws;                               // B*D
  float* scores = Q + (size_t)B_ * D_;              // B*H*S
  float* ff     = scores + (size_t)B_ * H_ * S_;    // B
  float* ss     = ff + B_;                          // S*B
  float* fs     = ss + (size_t)S_ * B_;             // S*B
  float* comb   = out + (size_t)B_ * C_;            // combined lives in d_out

  qgemm_kernel<<<dim3(B_ / BM, D_ / BN), 256, 0, stream>>>(feat, Wq, bq, Q);
  dots_kernel<<<dim3(B_ / 8), 256, 0, stream>>>(feat, sfeat, ff, ss, fs);
  kscore_kernel<<<dim3(B_ / BM, D_ / BN, S_), 256, 0, stream>>>(
      sfeat, semb, Wk, bk, Q, scores);
  fuse_kernel<<<dim3(B_ / 256), 256, 0, stream>>>(scores, ff, ss, fs, temp, attnw, comb);
  logits_kernel<<<dim3((int)((B_ * (size_t)C_) / (4 * 256))), 256, 0, stream>>>(
      logits, comb, out);
}
